// MultiheadAttention_21878563406366
// MI455X (gfx1250) — matrix operand order repaired and checked by reading, not compiled, not run on hardware
//
#include <hip/hip_runtime.h>
#include <hip/hip_bf16.h>

// ---------------------------------------------------------------------------
// MI455X (gfx1250) implementation.
// Shapes: B=2, N1=65536 (256x256), C=256, G=512, H=8, Dh=32, N2=16384.
// Heavy GEMMs: v_wmma_f32_16x16x32_f16 (wave32), f16 operands staged in LDS
// by the Tensor Data Mover (tensor_load_to_lds, double-buffered, TENSORcnt),
// fp32 accumulation, fused epilogues.  Small einsums are VALU kernels.
// ---------------------------------------------------------------------------

typedef __attribute__((ext_vector_type(16))) _Float16     v16h;
typedef __attribute__((ext_vector_type(8)))  float        v8f;
typedef __attribute__((ext_vector_type(4)))  unsigned int v4u;
typedef __attribute__((ext_vector_type(8)))  int          v8i;
typedef __attribute__((ext_vector_type(4)))  int          v4i;

#define ATT_EPS 1e-6f

#if __has_builtin(__builtin_amdgcn_tensor_load_to_lds)
#define HAVE_TDM 1
#endif

// ---- Tensor Data Mover: 2D tile global->LDS (ISA ch.8 D# layout) ----------
// dsize_code: 0=1B 1=2B 2=4B.  pad: after every 2^(pi+1) DWORDs insert pa+1 DWORDs.
__device__ __forceinline__ void tdm_load_2d(unsigned lds_off, const void* gptr,
                                            unsigned dsize_code,
                                            unsigned tensor_d0, unsigned tensor_d1,
                                            unsigned tile_d0, unsigned tile_d1,
                                            unsigned stride_d0,
                                            unsigned pad_interval, unsigned pad_amount)
{
#if defined(HAVE_TDM)
    const unsigned long long ga = (unsigned long long)gptr;
    v4u g0;
    g0.x = 1u;                                   // count=1: valid user descriptor
    g0.y = lds_off;                              // lds_addr (bytes)
    g0.z = (unsigned)ga;                         // global_addr[31:0] (tile start)
    g0.w = (unsigned)(ga >> 32) | (2u << 30);    // global_addr[56:32] | type=2
    v8i g1;
    g1[0] = (int)((dsize_code << 16) | (1u << 20) |          // data_size | pad_enable
                  (pad_interval << 22) | (pad_amount << 25));
    g1[1] = (int)((tensor_d0 & 0xFFFFu) << 16);              // tensor_dim0[15:0]
    g1[2] = (int)((tensor_d0 >> 16) | ((tensor_d1 & 0xFFFFu) << 16));
    g1[3] = (int)((tensor_d1 >> 16) | ((tile_d0 & 0xFFFFu) << 16));
    g1[4] = (int)(tile_d1 & 0xFFFFu);                        // tile_dim1 (tile_dim2=0)
    g1[5] = (int)stride_d0;                                  // tensor_dim0_stride[31:0]
    g1[6] = 0;
    g1[7] = 0;
    v4i z4 = (v4i)0;
#if __clang_major__ >= 23
    v8i z8 = (v8i)0;
    __builtin_amdgcn_tensor_load_to_lds(g0, g1, z4, z4, z8, 0);
#else
    __builtin_amdgcn_tensor_load_to_lds(g0, g1, z4, z4, 0);
#endif
#else
    (void)lds_off; (void)gptr; (void)dsize_code; (void)tensor_d0; (void)tensor_d1;
    (void)tile_d0; (void)tile_d1; (void)stride_d0; (void)pad_interval; (void)pad_amount;
#endif
}

__device__ __forceinline__ void tdm_wait0()
{
#if defined(HAVE_TDM)
    __builtin_amdgcn_s_wait_tensorcnt(0);
#endif
}

// -------------------------- weight f32 -> f16 ------------------------------
__global__ __launch_bounds__(256)
void cvt_f16_kernel(const float* __restrict__ src, _Float16* __restrict__ dst, int n)
{
    int base = blockIdx.x * 1024 + threadIdx.x;
    #pragma unroll
    for (int j = 0; j < 4; ++j) {
        int i = base + j * 256;
        if (i < n) dst[i] = (_Float16)src[i];
    }
}

// -------------------------- Haar pool kernel -------------------------------
// feat[b,c,y,x] = key[b, y*256+x, c].  high = 0.5*(-a-b-c+3d), low = 0.5*(a+b+c+d),
// s = high+low (= 2d).  One block per (b,m); thread = channel (coalesced).
__global__ __launch_bounds__(256)
void haar_kernel(const float* __restrict__ key,
                 float* __restrict__ high, float* __restrict__ low,
                 _Float16* __restrict__ s16)
{
    const int c  = threadIdx.x;
    const int bm = blockIdx.x;             // b*16384 + m
    const int b  = bm >> 14;
    const int m  = bm & 16383;
    const int i  = m >> 7, j = m & 127;
    const size_t base = ((size_t)b * 65536 + (size_t)(2 * i) * 256 + 2 * j) * 256 + c;
    const float a  = key[base];
    const float bb = key[base + 256];          // x+1
    const float cc = key[base + 65536];        // y+1
    const float dd = key[base + 65536 + 256];  // y+1, x+1
    const float hi = 0.5f * (-a - bb - cc + 3.0f * dd);
    const float lo = 0.5f * ( a + bb + cc +        dd);
    const size_t idx = (size_t)bm * 256 + c;
    high[idx] = hi;
    low[idx]  = lo;
    s16[idx]  = (_Float16)(hi + lo);
}

// ----------------------- pool partial reduction ----------------------------
__global__ __launch_bounds__(256)
void pool_partial_kernel(const _Float16* __restrict__ s16, float* __restrict__ partial)
{
    const int c     = threadIdx.x;
    const int b     = blockIdx.x >> 6;
    const int chunk = blockIdx.x & 63;
    size_t base = ((size_t)b * 16384 + (size_t)chunk * 256) * 256 + c;
    float acc = 0.0f;
    for (int m = 0; m < 256; ++m) acc += (float)s16[base + (size_t)m * 256];
    partial[(size_t)blockIdx.x * 256 + c] = acc;
}

// ----------------------------- glob kernel ---------------------------------
__global__ __launch_bounds__(256)
void glob_kernel(const float* __restrict__ partial,
                 const float* __restrict__ g1w, const float* __restrict__ g1b,
                 const float* __restrict__ g2w, const float* __restrict__ g2b,
                 float* __restrict__ glob)
{
    __shared__ float pool[256];
    __shared__ float hid[256];
    const int t = threadIdx.x;
    for (int b = 0; b < 2; ++b) {
        float acc = 0.0f;
        for (int ch = 0; ch < 64; ++ch) acc += partial[(size_t)(b * 64 + ch) * 256 + t];
        pool[t] = acc * (1.0f / 16384.0f);
        __syncthreads();
        float h = g1b[t];
        for (int i = 0; i < 256; ++i) h = __builtin_fmaf(pool[i], g1w[i * 256 + t], h);
        hid[t] = fmaxf(h, 0.0f);
        __syncthreads();
        float g = g2b[t];
        for (int i = 0; i < 256; ++i) g = __builtin_fmaf(hid[i], g2w[i * 256 + t], g);
        glob[b * 256 + t] = g;
        __syncthreads();
    }
}

// ------------------------- WMMA GEMM (N = 256) -----------------------------
// out[M x 256] = epi(A[M x KDIM] @ W[KDIM x 256] + bias),  W in f16.
// Block: 256 threads = 8 waves; block tile = 32 rows x 256 cols.
// Wave (rg,cg): rows rg*16..+15, cols cg*64..+63 as 4 WMMA 16x16 tiles.
// Tiles staged in LDS by TDM (wave 0 issues, TENSORcnt-waited), double-buffered
// so the DMA of step k+1 overlaps the WMMA of step k.  TDM pad_amount adds the
// LDS row padding (As: 16DW+1DW, Bs: 128DW+4DW) for bank-conflict avoidance.
enum { EPI_LIN = 0, EPI_RELU = 1, EPI_ELU1 = 2, EPI_FRE = 3 };

template <int KDIM, int EPI, bool A16>
__global__ __launch_bounds__(256)
void gemm256_kernel(const void* __restrict__ Aptr, const _Float16* __restrict__ W,
                    const float* __restrict__ bias, _Float16* __restrict__ out,
                    const float* __restrict__ glob, const float* __restrict__ high,
                    const float* __restrict__ low)
{
    constexpr int ASTRIDE = A16 ? 34 : 33;                  // halves / floats per row
    constexpr int ABYTES  = A16 ? (32 * 34 * 2) : (32 * 33 * 4);
    __shared__ __align__(16) char     AsRaw[2][ABYTES];     // 32 rows x 32 k (padded)
    __shared__ __align__(16) _Float16 Bs[2][32][264];       // 32 k x 256 n (padded)

    const int tid  = threadIdx.x;
    const int wid  = tid >> 5;
    const int lane = tid & 31;
    const int rg   = wid >> 2;   // row group (0..1)
    const int cg   = wid & 3;    // col group (0..3)
    const long row0  = (long)blockIdx.x * 32;
    const long Mrows = (long)gridDim.x * 32;

    v8f acc[4] = {v8f{}, v8f{}, v8f{}, v8f{}};

#if defined(HAVE_TDM)
    auto issue_tiles = [&](int k0, int p) {
        const size_t esz = A16 ? 2 : 4;
        const char* abase = (const char*)Aptr + ((size_t)row0 * KDIM + k0) * esz;
        // A tile: 32 rows x 32 elems; LDS rows padded by 1 DWORD.
        tdm_load_2d((unsigned)(unsigned long long)&AsRaw[p][0], abase,
                    A16 ? 1u : 2u, (unsigned)KDIM, (unsigned)Mrows,
                    32u, 32u, (unsigned)KDIM,
                    A16 ? 3u : 4u /* 16 / 32 DW */, 0u /* +1 DW */);
        // W tile: 32 rows x 256 f16; LDS rows padded by 4 DWORDs (264 halves).
        tdm_load_2d((unsigned)(unsigned long long)&Bs[p][0][0], W + (size_t)k0 * 256,
                    1u, 256u, (unsigned)KDIM, 256u, 32u, 256u,
                    6u /* 128 DW */, 3u /* +4 DW */);
    };
    if (wid == 0) { issue_tiles(0, 0); tdm_wait0(); }
    __syncthreads();
#else
    auto stage_tiles = [&](int k0, int p) {
        if (A16) {
            const _Float16* A = (const _Float16*)Aptr;
            _Float16* As = (_Float16*)AsRaw[p];
            #pragma unroll
            for (int i = 0; i < 4; ++i) {
                int e = tid + i * 256, r = e >> 5, kk = e & 31;
                As[r * ASTRIDE + kk] = A[(size_t)(row0 + r) * KDIM + k0 + kk];
            }
        } else {
            const float* A = (const float*)Aptr;
            float* As = (float*)AsRaw[p];
            #pragma unroll
            for (int i = 0; i < 4; ++i) {
                int e = tid + i * 256, r = e >> 5, kk = e & 31;
                As[r * ASTRIDE + kk] = A[(size_t)(row0 + r) * KDIM + k0 + kk];
            }
        }
        #pragma unroll 8
        for (int i = 0; i < 32; ++i)
            Bs[p][i][tid] = W[(size_t)(k0 + i) * 256 + tid];
    };
    stage_tiles(0, 0);
    __syncthreads();
#endif

    int p = 0;
    for (int k0 = 0; k0 < KDIM; k0 += 32) {
        const bool has_next = (k0 + 32 < KDIM);
#if defined(HAVE_TDM)
        if (has_next && wid == 0) issue_tiles(k0 + 32, p ^ 1);   // overlap DMA
#else
        if (has_next) stage_tiles(k0 + 32, p ^ 1);
#endif
        // ---- A fragment: 16x32 f16 layout (ISA 7.12.2) ----
        // lane<16: M=lane, K in {0..7,16..23}; lane>=16: M=lane-16, K+8
        v16h afrag;
        const int am    = (lane & 15) + rg * 16;
        const int kbase = (lane < 16) ? 0 : 8;
        #pragma unroll
        for (int h = 0; h < 16; ++h) {
            int v = h >> 1, q = h & 1;
            int kk = ((v < 4) ? (2 * v + q) : (16 + 2 * (v - 4) + q)) + kbase;
            if (A16) afrag[h] = ((const _Float16*)AsRaw[p])[am * ASTRIDE + kk];
            else     afrag[h] = (_Float16)((const float*)AsRaw[p])[am * ASTRIDE + kk];
        }
        // ---- 4 B fragments + WMMA; B layout: lane%16 = N, halves = K (same order as A) ----
        #pragma unroll
        for (int t = 0; t < 4; ++t) {
            v16h bfrag;
            #pragma unroll
            for (int h = 0; h < 16; ++h)
                bfrag[h] = Bs[p][((h < 8) ? h : (h + 8)) + kbase][cg * 64 + t * 16 + (lane & 15)];
            acc[t] = __builtin_amdgcn_wmma_f32_16x16x32_f16(
                false, afrag, false, bfrag, (short)0, acc[t], false, false);
        }
        __syncthreads();                          // all waves done reading buffer p
#if defined(HAVE_TDM)
        if (has_next) {
            if (wid == 0) tdm_wait0();            // buffer p^1 fully landed
            __syncthreads();                      // publish to all waves
        }
#endif
        p ^= 1;
    }

    // ---- epilogue + store (C/D layout: lane 0-15 -> M=r, 16-31 -> M=r+8) ----
    const int  colb = cg * 64 + (lane & 15);
    const long rowb = row0 + rg * 16 + ((lane >> 4) << 3);
    float bv[4] = {0.0f, 0.0f, 0.0f, 0.0f};
    if (bias) {
        #pragma unroll
        for (int t = 0; t < 4; ++t) bv[t] = bias[colb + t * 16];
    }
    #pragma unroll
    for (int t = 0; t < 4; ++t) {
        const int col = colb + t * 16;
        #pragma unroll
        for (int r = 0; r < 8; ++r) {
            const long row = rowb + r;
            float x = acc[t][r] + bv[t];
            if (EPI == EPI_RELU) {
                x = fmaxf(x, 0.0f);
            } else if (EPI == EPI_ELU1) {
                x = (x > 0.0f) ? (x + 1.0f) : __expf(x);   // elu(x)+1
            } else if (EPI == EPI_FRE) {
                const int b = (int)(row >> 14);            // rows per batch = 16384
                const float wei = 1.0f / (1.0f + __expf(-(x + glob[b * 256 + col])));
                const size_t idx = (size_t)row * 256 + col;
                x = wei * high[idx] + low[idx];
            }
            out[(size_t)row * 256 + col] = (_Float16)x;
        }
    }
}

// ------------------------- KV / Ksum reduction -----------------------------
// KV[b,h,d,v] = sum_s K[b,s,h,d]*vh[b,s,h,v];  Ksum[b,h,d] = sum_s K.
__global__ __launch_bounds__(256)
void kv_kernel(const _Float16* __restrict__ Km, const _Float16* __restrict__ Vm,
               float* __restrict__ KV, float* __restrict__ Ksum)
{
    __shared__ float Ks[64][33];
    __shared__ float Vs[64][33];
    const int b  = blockIdx.z, h = blockIdx.y;
    const int s0 = blockIdx.x * 2048;
    const int tid = threadIdx.x;
    const int v  = tid & 31;
    const int dg = tid >> 5;       // handles d = dg*4 .. dg*4+3
    float acc[4] = {0.f, 0.f, 0.f, 0.f};
    float ks = 0.0f;

    for (int st = 0; st < 2048; st += 64) {
        __syncthreads();
        #pragma unroll
        for (int i = 0; i < 8; ++i) {
            int e = tid + i * 256;
            int sl = e >> 5, d = e & 31;
            size_t g = ((size_t)b * 16384 + s0 + st + sl) * 256 + h * 32 + d;
            Ks[sl][d] = (float)Km[g];
            Vs[sl][d] = (float)Vm[g];
        }
        __syncthreads();
        for (int s = 0; s < 64; ++s) {
            float vv = Vs[s][v];
            #pragma unroll
            for (int jj = 0; jj < 4; ++jj)
                acc[jj] = __builtin_fmaf(Ks[s][dg * 4 + jj], vv, acc[jj]);
        }
        if (tid < 32)
            for (int s = 0; s < 64; ++s) ks += Ks[s][tid];
    }
    const int hb = b * 8 + h;
    #pragma unroll
    for (int jj = 0; jj < 4; ++jj)
        atomicAdd(&KV[((size_t)hb * 32 + dg * 4 + jj) * 32 + v], acc[jj]);
    if (tid < 32) atomicAdd(&Ksum[hb * 32 + tid], ks);
}

// ----------------------------- final kernel --------------------------------
// out[b,l,h,v] = (sum_d Q[b,l,h,d]*KV[b,h,d,v]) / (Q . Ksum + eps)
__global__ __launch_bounds__(256)
void final_kernel(const _Float16* __restrict__ Qm, const float* __restrict__ KV,
                  const float* __restrict__ Ksum, float* __restrict__ out)
{
    __shared__ float KVs[8][32][33];
    __shared__ float Kss[8][32];
    const int b  = blockIdx.y;
    const int l0 = blockIdx.x * 32;
    const int tid = threadIdx.x;

    #pragma unroll
    for (int i = 0; i < 32; ++i) {
        int e = tid + i * 256;                 // 0..8191
        int hh = e >> 10, dd = (e >> 5) & 31, vv = e & 31;
        KVs[hh][dd][vv] = KV[(size_t)(b * 8 + hh) * 1024 + dd * 32 + vv];
    }
    {
        int hh = tid >> 5, dd = tid & 31;
        Kss[hh][dd] = Ksum[(b * 8 + hh) * 32 + dd];
    }
    __syncthreads();

    const int l = l0 + (tid >> 3);
    const int h = tid & 7;
    const size_t base = ((size_t)b * 65536 + l) * 256 + h * 32;

    float q[32];
    #pragma unroll
    for (int d = 0; d < 32; ++d) q[d] = (float)Qm[base + d];

    float z = ATT_EPS;
    #pragma unroll
    for (int d = 0; d < 32; ++d) z = __builtin_fmaf(q[d], Kss[h][d], z);
    z = 1.0f / z;

    float* o = out + base;
    for (int vv = 0; vv < 32; ++vv) {
        float a = 0.0f;
        #pragma unroll
        for (int d = 0; d < 32; ++d) a = __builtin_fmaf(q[d], KVs[h][d][vv], a);
        o[vv] = a * z;
    }
}

// ------------------------------- launcher ----------------------------------
extern "C" void kernel_launch(void* const* d_in, const int* in_sizes, int n_in,
                              void* d_out, int out_size, void* d_ws, size_t ws_size,
                              hipStream_t stream)
{
    (void)in_sizes; (void)n_in; (void)out_size; (void)ws_size;
    const float* query    = (const float*)d_in[0];
    const float* key      = (const float*)d_in[1];
    // d_in[2] (value) only supplies shapes in the reference — unused.
    const float* mheads1w = (const float*)d_in[3];
    const float* q_w = (const float*)d_in[4],  *q_b = (const float*)d_in[5];
    const float* k_w = (const float*)d_in[6],  *k_b = (const float*)d_in[7];
    const float* v_w = (const float*)d_in[8],  *v_b = (const float*)d_in[9];
    const float* l1w = (const float*)d_in[10], *l1b = (const float*)d_in[11];
    const float* l2w = (const float*)d_in[12], *l2b = (const float*)d_in[13];
    const float* g1w = (const float*)d_in[14], *g1b = (const float*)d_in[15];
    const float* g2w = (const float*)d_in[16], *g2b = (const float*)d_in[17];
    float* out = (float*)d_out;

    // ---- workspace layout ----
    char* ws = (char*)d_ws;
    size_t off = 0;
    auto alloc = [&](size_t bytes) { size_t o = off; off += (bytes + 255) & ~(size_t)255; return o; };
    const size_t n2c = (size_t)2 * 16384 * 256;   // pooled activations
    const size_t n1c = (size_t)2 * 65536 * 256;   // full-seq activations
    float*    high  = (float*)   (ws + alloc(n2c * 4));
    float*    low   = (float*)   (ws + alloc(n2c * 4));
    _Float16* s16   = (_Float16*)(ws + alloc(n2c * 2));
    _Float16* hid16 = (_Float16*)(ws + alloc(n2c * 2));
    _Float16* fre16 = (_Float16*)(ws + alloc(n2c * 2));
    _Float16* K16   = (_Float16*)(ws + alloc(n2c * 2));
    _Float16* V16   = (_Float16*)(ws + alloc(n2c * 2));
    _Float16* qg16  = (_Float16*)(ws + alloc(n1c * 2));
    _Float16* Q16   = (_Float16*)(ws + alloc(n1c * 2));
    // f16 weight panels
    _Float16* wl1 = (_Float16*)(ws + alloc((size_t)65536 * 2));
    _Float16* wl2 = (_Float16*)(ws + alloc((size_t)65536 * 2));
    _Float16* wk  = (_Float16*)(ws + alloc((size_t)65536 * 2));
    _Float16* wv  = (_Float16*)(ws + alloc((size_t)65536 * 2));
    _Float16* wq  = (_Float16*)(ws + alloc((size_t)65536 * 2));
    _Float16* wmh = (_Float16*)(ws + alloc((size_t)131072 * 2));
    float*    part = (float*)(ws + alloc((size_t)128 * 256 * 4));
    float*    glob = (float*)(ws + alloc((size_t)2 * 256 * 4));
    float*    KV   = (float*)(ws + alloc((size_t)2 * 8 * 32 * 32 * 4)); // + Ksum adjacent
    float*    Ksum = (float*)(ws + alloc((size_t)2 * 8 * 32 * 4));

    // KV/Ksum are atomically accumulated -> zero every launch (capturable).
    hipMemsetAsync(KV, 0, (size_t)2 * 8 * 32 * 32 * 4 + 2048, stream);

    // 0. weight panels f32 -> f16 (tiny, once per launch)
    cvt_f16_kernel<<<64, 256, 0, stream>>>(l1w, wl1, 65536);
    cvt_f16_kernel<<<64, 256, 0, stream>>>(l2w, wl2, 65536);
    cvt_f16_kernel<<<64, 256, 0, stream>>>(k_w, wk, 65536);
    cvt_f16_kernel<<<64, 256, 0, stream>>>(v_w, wv, 65536);
    cvt_f16_kernel<<<64, 256, 0, stream>>>(q_w, wq, 65536);
    cvt_f16_kernel<<<128, 256, 0, stream>>>(mheads1w, wmh, 131072);

    // 1. Haar pool -> high/low (f32), s (f16)
    haar_kernel<<<32768, 256, 0, stream>>>(key, high, low, s16);
    // 2. pool mean + glob MLP (tiny)
    pool_partial_kernel<<<128, 256, 0, stream>>>(s16, part);
    glob_kernel<<<1, 256, 0, stream>>>(part, g1w, g1b, g2w, g2b, glob);
    // 3. hidden = relu(s @ l1 + b1)
    gemm256_kernel<256, EPI_RELU, true><<<1024, 256, 0, stream>>>(
        s16, wl1, l1b, hid16, nullptr, nullptr, nullptr);
    // 4. fre = sigmoid(hidden @ l2 + b2 + glob) * high + low
    gemm256_kernel<256, EPI_FRE, true><<<1024, 256, 0, stream>>>(
        hid16, wl2, l2b, fre16, glob, high, low);
    // 5. K = elu(fre @ k_w + k_b) + 1 ;  vh = fre @ v_w + v_b
    gemm256_kernel<256, EPI_ELU1, true><<<1024, 256, 0, stream>>>(
        fre16, wk, k_b, K16, nullptr, nullptr, nullptr);
    gemm256_kernel<256, EPI_LIN, true><<<1024, 256, 0, stream>>>(
        fre16, wv, v_b, V16, nullptr, nullptr, nullptr);
    // 6. q_guid = query @ mheads1_w   (K=512, f32 A, no bias)
    gemm256_kernel<512, EPI_LIN, false><<<4096, 256, 0, stream>>>(
        query, wmh, nullptr, qg16, nullptr, nullptr, nullptr);
    // 7. Q = elu(q_guid @ q_w + q_b) + 1
    gemm256_kernel<256, EPI_ELU1, true><<<4096, 256, 0, stream>>>(
        qg16, wq, q_b, Q16, nullptr, nullptr, nullptr);
    // 8. KV = K^T vh, Ksum = sum_s K   (the /S and *S in the reference cancel)
    kv_kernel<<<dim3(8, 8, 2), 256, 0, stream>>>(K16, V16, KV, Ksum);
    // 9. out = (Q @ KV) / (Q . Ksum + eps)
    final_kernel<<<dim3(2048, 2), 256, 0, stream>>>(Q16, KV, Ksum, out);
}
